// MoE_7241314861588
// MI455X (gfx1250) — matrix operand order repaired and checked by reading, not compiled, not run on hardware
//
#include <hip/hip_runtime.h>
#include <hip/hip_bf16.h>

// ---------------------------------------------------------------------------
// MoE (top-2 routed + shared expert, SwiGLU) for gfx1250 / MI455X.
// Core math on v_wmma_f32_16x16x32_bf16, wave32, 8 waves/block, 128x64 tiles.
// LDS double buffering fed by GLOBAL_LOAD_ASYNC_TO_LDS_B128 (ASYNCcnt) when
// the toolchain exposes the builtin; register-staged ds_store path otherwise.
// ---------------------------------------------------------------------------

typedef unsigned int   v4u     __attribute__((ext_vector_type(4)));
typedef unsigned short u16x4   __attribute__((ext_vector_type(4)));
typedef __bf16         bf16x16 __attribute__((ext_vector_type(16)));
typedef float          f32x8   __attribute__((ext_vector_type(8)));
typedef int            i32x4   __attribute__((ext_vector_type(4)));

#define E_EXP 16
#define TOPK  2
#define TM    128   // block tile M
#define TN    64    // block tile N
#define TK    32    // k-step (WMMA K)
#define XP    40    // LDS pitch (halves): 32 + 8 pad -> 80B rows, conflict-free
#define WP    40

#if defined(__gfx1250__) && __has_builtin(__builtin_amdgcn_global_load_async_to_lds_b128)
#define USE_ASYNC 1
#else
#define USE_ASYNC 0
#endif

#if USE_ASYNC
typedef __attribute__((address_space(1))) i32x4 glb_i32x4;   // global int4
typedef __attribute__((address_space(3))) i32x4 lds_i32x4;   // LDS int4
// 16B per-lane async copy global -> LDS. AS pointers are reconstructed from
// integer addresses: global generic VA == AS1 VA; LDS generic VA low 32 bits
// are the LDS byte offset (aperture rules, ISA 10.2).
static __device__ __forceinline__ void async_cp16(const unsigned short* g,
                                                  unsigned short* l) {
  unsigned long long ga = (unsigned long long)g;
  unsigned           la = (unsigned)(unsigned long long)l;
  __builtin_amdgcn_global_load_async_to_lds_b128((glb_i32x4*)ga, (lds_i32x4*)la,
                                                 0, 0);
}
static __device__ __forceinline__ void wait_async0() {
#if __has_builtin(__builtin_amdgcn_s_wait_asynccnt)
  __builtin_amdgcn_s_wait_asynccnt(0);
#else
  asm volatile("s_wait_asynccnt 0x0" ::: "memory");
#endif
}
#endif

static __device__ __forceinline__ float bf2f(unsigned short u) {
  return __uint_as_float(((unsigned)u) << 16);
}
static __device__ __forceinline__ unsigned short f2bf(float f) {
  unsigned u = __float_as_uint(f);
  u += 0x7FFFu + ((u >> 16) & 1u);   // round-to-nearest-even
  return (unsigned short)(u >> 16);
}

// A fragment: 16x32 bf16. Lane half 0: K 0-7 & 16-23; half 1: K 8-15 & 24-31.
static __device__ __forceinline__ bf16x16 frag_a(const unsigned short* p) {
  union { v4u u[2]; bf16x16 f; } t;
  t.u[0] = *(const v4u*)(p);
  t.u[1] = *(const v4u*)(p + 16);
  return t.f;
}
// B fragment: lane (l&15)=N column. Lane half 0: K 0-7 & 16-23; half 1: K 8-15 & 24-31.
static __device__ __forceinline__ bf16x16 frag_b(const unsigned short* p) {
  union { v4u u[2]; bf16x16 f; } t;
  t.u[0] = *(const v4u*)(p);
  t.u[1] = *(const v4u*)(p + 16);
  return t.f;
}
static __device__ __forceinline__ f32x8 wmma_bf16(bf16x16 a, bf16x16 b, f32x8 c) {
  return __builtin_amdgcn_wmma_f32_16x16x32_bf16(false, a, false, b, (short)0, c,
                                                 false, false);
}

// ---------------------------------------------------------------------------
// Gate: scores = sigmoid(x @ gate_w^T), top-2, normalized; histogram counts.
// ---------------------------------------------------------------------------
__global__ __launch_bounds__(128)
void moe_gate_kernel(const unsigned short* __restrict__ X,
                     const float* __restrict__ GW,
                     int* __restrict__ top2i, float* __restrict__ top2w,
                     int* __restrict__ counts, int T, int D)
{
  __shared__ float part[128];
  __shared__ float sc[E_EXP];
  const int t  = blockIdx.x;
  const int tid = threadIdx.x;
  const int e  = tid & (E_EXP - 1);
  const int sl = tid >> 4;                  // 8 k-slices per expert
  const unsigned short* xp = X + (size_t)t * D;
  const float* gp = GW + (size_t)e * D;
  float s = 0.f;
  for (int k = sl; k < D; k += 8) s += bf2f(xp[k]) * gp[k];
  part[tid] = s;
  __syncthreads();
  if (tid < E_EXP) {
    float v = 0.f;
#pragma unroll
    for (int j = 0; j < 8; ++j) v += part[j * E_EXP + tid];
    sc[tid] = 1.0f / (1.0f + __expf(-v));
  }
  __syncthreads();
  if (tid == 0) {
    int i0 = 0; float v0 = sc[0];
    for (int k = 1; k < E_EXP; ++k) if (sc[k] > v0) { v0 = sc[k]; i0 = k; }
    int i1 = (i0 == 0) ? 1 : 0; float v1 = sc[i1];
    for (int k = 0; k < E_EXP; ++k)
      if (k != i0 && sc[k] > v1) { v1 = sc[k]; i1 = k; }
    float inv = 1.0f / (v0 + v1);
    top2i[t * 2 + 0] = i0; top2w[t * 2 + 0] = v0 * inv;
    top2i[t * 2 + 1] = i1; top2w[t * 2 + 1] = v1 * inv;
    atomicAdd(&counts[i0], 1);
    atomicAdd(&counts[i1], 1);
  }
}

__global__ void moe_init_kernel(int* counts) {
  if (threadIdx.x < E_EXP) counts[threadIdx.x] = 0;
}

__global__ void moe_scan_kernel(const int* __restrict__ counts,
                                int* __restrict__ offsets,
                                int* __restrict__ cursor) {
  if (threadIdx.x == 0) {
    int o = 0;
    for (int e = 0; e < E_EXP; ++e) { offsets[e] = o; cursor[e] = o; o += counts[e]; }
  }
}

__global__ void moe_scatter_kernel(const int* __restrict__ top2i,
                                   const float* __restrict__ top2w,
                                   int* __restrict__ cursor,
                                   int* __restrict__ tokidx,
                                   float* __restrict__ tokw, int T) {
  int t = blockIdx.x * blockDim.x + threadIdx.x;
  if (t >= T) return;
#pragma unroll
  for (int j = 0; j < TOPK; ++j) {
    int e = top2i[t * 2 + j];
    int slot = atomicAdd(&cursor[e], 1);
    tokidx[slot] = t;
    tokw[slot] = top2w[t * 2 + j];
  }
}

// ---------------------------------------------------------------------------
// Up-projection (fused SwiGLU): H[m,n] = silu(X@W1^T)[m,n] * (X@W3^T)[m,n].
// routed: grid.z = expert; token rows gathered via tokidx; early-exit on count.
// ---------------------------------------------------------------------------
__global__ __launch_bounds__(256)
void moe_up_kernel(const unsigned short* __restrict__ X,
                   const unsigned short* __restrict__ W1,
                   const unsigned short* __restrict__ W3,
                   unsigned short* __restrict__ H,
                   const int* __restrict__ tokidx,
                   const int* __restrict__ counts,
                   const int* __restrict__ offsets,
                   int T, int N, int K, int routed)
{
  __shared__ unsigned short Xs[2][TM * XP];
  __shared__ unsigned short W1s[2][TN * WP];
  __shared__ unsigned short W3s[2][TN * WP];

  int cnt = T, base = 0;
  const unsigned short* w1p = W1;
  const unsigned short* w3p = W3;
  if (routed) {
    const int e = blockIdx.z;
    cnt = counts[e];
    if ((int)blockIdx.y * TM >= cnt) return;
    base = offsets[e];
    const size_t wo = (size_t)e * (size_t)N * (size_t)K;
    w1p = W1 + wo; w3p = W3 + wo;
  }
  const int mbase = blockIdx.y * TM;
  const int nbase = blockIdx.x * TN;
  const int tid  = threadIdx.x;
  const int lane = tid & 31;
  const int wave = tid >> 5;
  const int wm   = wave >> 1;          // 0..3 : 32-row slab
  const int wn   = wave & 1;           // 0..1 : 32-col slab
  const int half = lane >> 4;
  const int l16  = lane & 15;

  // Staging map: X tile 128x32 -> 16 contiguous halves per thread (2 x b128)
  const int xr = tid >> 1;
  const int xc = (tid & 1) * 16;
  int tok;
  if (routed) {
    int m = mbase + xr; if (m > cnt - 1) m = cnt - 1;
    tok = tokidx[base + m];
  } else {
    tok = mbase + xr;
  }
  const unsigned short* xg = X + (size_t)tok * (size_t)K + xc;

  // W tiles 64x32 -> 8 halves per thread per matrix (1 x b128 each)
  const int wr = tid >> 2;
  const int wc = (tid & 3) * 8;
  const unsigned short* w1g = w1p + (size_t)(nbase + wr) * (size_t)K + wc;
  const unsigned short* w3g = w3p + (size_t)(nbase + wr) * (size_t)K + wc;

  const int nk = K / TK;

  const f32x8 z = {0.f,0.f,0.f,0.f,0.f,0.f,0.f,0.f};
  f32x8 acc1[2][2], acc3[2][2];
#pragma unroll
  for (int i = 0; i < 2; ++i)
#pragma unroll
    for (int j = 0; j < 2; ++j) { acc1[i][j] = z; acc3[i][j] = z; }

#if USE_ASYNC
  async_cp16(xg,      &Xs[0][xr * XP + xc]);
  async_cp16(xg + 8,  &Xs[0][xr * XP + xc + 8]);
  async_cp16(w1g,     &W1s[0][wr * WP + wc]);
  async_cp16(w3g,     &W3s[0][wr * WP + wc]);
  wait_async0();
  __syncthreads();
#else
  v4u rx0, rx1, rw1, rw3;
  rx0 = *(const v4u*)(xg);
  rx1 = *(const v4u*)(xg + 8);
  rw1 = *(const v4u*)(w1g);
  rw3 = *(const v4u*)(w3g);
  *(v4u*)&Xs[0][xr * XP + xc]     = rx0;
  *(v4u*)&Xs[0][xr * XP + xc + 8] = rx1;
  *(v4u*)&W1s[0][wr * WP + wc]    = rw1;
  *(v4u*)&W3s[0][wr * WP + wc]    = rw3;
  __syncthreads();
#endif

  for (int kt = 0; kt < nk; ++kt) {
    const int cur = kt & 1;
    const int nxt = cur ^ 1;
    if (kt + 1 < nk) {                         // prefetch next k-tile
      const int k0 = (kt + 1) * TK;
#if USE_ASYNC
      // Safe: barrier at end of iteration kt-1 guarantees every wave finished
      // reading buffer `nxt`; DMA overlaps the WMMA work below.
      async_cp16(xg + k0,     &Xs[nxt][xr * XP + xc]);
      async_cp16(xg + k0 + 8, &Xs[nxt][xr * XP + xc + 8]);
      async_cp16(w1g + k0,    &W1s[nxt][wr * WP + wc]);
      async_cp16(w3g + k0,    &W3s[nxt][wr * WP + wc]);
#else
      rx0 = *(const v4u*)(xg + k0);
      rx1 = *(const v4u*)(xg + k0 + 8);
      rw1 = *(const v4u*)(w1g + k0);
      rw3 = *(const v4u*)(w3g + k0);
#endif
    }
    bf16x16 a[2];
#pragma unroll
    for (int i = 0; i < 2; ++i)
      a[i] = frag_a(&Xs[cur][(wm * 32 + i * 16 + l16) * XP + half * 8]);
#pragma unroll
    for (int j = 0; j < 2; ++j) {
      const bf16x16 b1 = frag_b(&W1s[cur][(wn * 32 + j * 16 + l16) * WP + half * 8]);
      const bf16x16 b3 = frag_b(&W3s[cur][(wn * 32 + j * 16 + l16) * WP + half * 8]);
#pragma unroll
      for (int i = 0; i < 2; ++i) {
        acc1[i][j] = wmma_bf16(a[i], b1, acc1[i][j]);
        acc3[i][j] = wmma_bf16(a[i], b3, acc3[i][j]);
      }
    }
    if (kt + 1 < nk) {
#if USE_ASYNC
      wait_async0();
      __syncthreads();
#else
      __syncthreads();
      *(v4u*)&Xs[nxt][xr * XP + xc]     = rx0;
      *(v4u*)&Xs[nxt][xr * XP + xc + 8] = rx1;
      *(v4u*)&W1s[nxt][wr * WP + wc]    = rw1;
      *(v4u*)&W3s[nxt][wr * WP + wc]    = rw3;
      __syncthreads();
#endif
    }
  }

  // Epilogue: SwiGLU in f32 (fast rcp; output is bf16 so 1-ulp rcp is exact
  // enough), store bf16 rows of H.
#pragma unroll
  for (int i = 0; i < 2; ++i) {
#pragma unroll
    for (int j = 0; j < 2; ++j) {
      const int m0 = mbase + wm * 32 + i * 16 + half * 8;
      const int nc = nbase + wn * 32 + j * 16 + l16;
#pragma unroll
      for (int r = 0; r < 8; ++r) {
        const int m = m0 + r;
        if (m < cnt) {
          const float av = acc1[i][j][r];
          const float bv = acc3[i][j][r];
          const float s  = av * __builtin_amdgcn_rcpf(1.0f + __expf(-av)); // silu
          H[(size_t)(base + m) * (size_t)N + nc] = f2bf(s * bv);
        }
      }
    }
  }
}

// ---------------------------------------------------------------------------
// Down-projection: out = H @ W2^T. Shared path stores f32 Y; routed path does
// Y[token] += combine_weight * out via global f32 atomics.
// ---------------------------------------------------------------------------
__global__ __launch_bounds__(256)
void moe_down_kernel(const unsigned short* __restrict__ Hc,
                     const unsigned short* __restrict__ W2,
                     float* __restrict__ Y,
                     const int* __restrict__ tokidx,
                     const float* __restrict__ tokw,
                     const int* __restrict__ counts,
                     const int* __restrict__ offsets,
                     int T, int N, int K, int routed)
{
  __shared__ unsigned short Hs[2][TM * XP];
  __shared__ unsigned short Ws[2][TN * WP];

  int cnt = T, base = 0;
  const unsigned short* wp = W2;
  if (routed) {
    const int e = blockIdx.z;
    cnt = counts[e];
    if ((int)blockIdx.y * TM >= cnt) return;
    base = offsets[e];
    wp = W2 + (size_t)e * (size_t)N * (size_t)K;
  }
  const int mbase = blockIdx.y * TM;
  const int nbase = blockIdx.x * TN;
  const int tid  = threadIdx.x;
  const int lane = tid & 31;
  const int wave = tid >> 5;
  const int wm   = wave >> 1;
  const int wn   = wave & 1;
  const int half = lane >> 4;
  const int l16  = lane & 15;

  const int xr = tid >> 1;
  const int xc = (tid & 1) * 16;
  int mrow = mbase + xr;
  if (mrow > cnt - 1) mrow = cnt - 1;
  const unsigned short* hg = Hc + (size_t)(base + mrow) * (size_t)K + xc;

  const int wr = tid >> 2;
  const int wc = (tid & 3) * 8;
  const unsigned short* wg = wp + (size_t)(nbase + wr) * (size_t)K + wc;

  const int nk = K / TK;

  const f32x8 z = {0.f,0.f,0.f,0.f,0.f,0.f,0.f,0.f};
  f32x8 acc[2][2];
#pragma unroll
  for (int i = 0; i < 2; ++i)
#pragma unroll
    for (int j = 0; j < 2; ++j) acc[i][j] = z;

#if USE_ASYNC
  async_cp16(hg,     &Hs[0][xr * XP + xc]);
  async_cp16(hg + 8, &Hs[0][xr * XP + xc + 8]);
  async_cp16(wg,     &Ws[0][wr * WP + wc]);
  wait_async0();
  __syncthreads();
#else
  v4u rh0, rh1, rw;
  rh0 = *(const v4u*)(hg);
  rh1 = *(const v4u*)(hg + 8);
  rw  = *(const v4u*)(wg);
  *(v4u*)&Hs[0][xr * XP + xc]     = rh0;
  *(v4u*)&Hs[0][xr * XP + xc + 8] = rh1;
  *(v4u*)&Ws[0][wr * WP + wc]     = rw;
  __syncthreads();
#endif

  for (int kt = 0; kt < nk; ++kt) {
    const int cur = kt & 1;
    const int nxt = cur ^ 1;
    if (kt + 1 < nk) {
      const int k0 = (kt + 1) * TK;
#if USE_ASYNC
      async_cp16(hg + k0,     &Hs[nxt][xr * XP + xc]);
      async_cp16(hg + k0 + 8, &Hs[nxt][xr * XP + xc + 8]);
      async_cp16(wg + k0,     &Ws[nxt][wr * WP + wc]);
#else
      rh0 = *(const v4u*)(hg + k0);
      rh1 = *(const v4u*)(hg + k0 + 8);
      rw  = *(const v4u*)(wg + k0);
#endif
    }
    bf16x16 a[2];
#pragma unroll
    for (int i = 0; i < 2; ++i)
      a[i] = frag_a(&Hs[cur][(wm * 32 + i * 16 + l16) * XP + half * 8]);
#pragma unroll
    for (int j = 0; j < 2; ++j) {
      const bf16x16 b = frag_b(&Ws[cur][(wn * 32 + j * 16 + l16) * WP + half * 8]);
#pragma unroll
      for (int i = 0; i < 2; ++i)
        acc[i][j] = wmma_bf16(a[i], b, acc[i][j]);
    }
    if (kt + 1 < nk) {
#if USE_ASYNC
      wait_async0();
      __syncthreads();
#else
      __syncthreads();
      *(v4u*)&Hs[nxt][xr * XP + xc]     = rh0;
      *(v4u*)&Hs[nxt][xr * XP + xc + 8] = rh1;
      *(v4u*)&Ws[nxt][wr * WP + wc]     = rw;
      __syncthreads();
#endif
    }
  }

#pragma unroll
  for (int i = 0; i < 2; ++i) {
#pragma unroll
    for (int j = 0; j < 2; ++j) {
      const int m0 = mbase + wm * 32 + i * 16 + half * 8;
      const int nc = nbase + wn * 32 + j * 16 + l16;
#pragma unroll
      for (int r = 0; r < 8; ++r) {
        const int m = m0 + r;
        if (m < cnt) {
          const float v = acc[i][j][r];
          if (routed) {
            const int slot = base + m;
            atomicAdd(&Y[(size_t)tokidx[slot] * (size_t)N + nc], v * tokw[slot]);
          } else {
            Y[(size_t)m * (size_t)N + nc] = v;
          }
        }
      }
    }
  }
}

// ---------------------------------------------------------------------------
// Finalize: f32 accumulator -> bf16 output.
// ---------------------------------------------------------------------------
__global__ __launch_bounds__(256)
void moe_finalize_kernel(const float* __restrict__ Y,
                         unsigned short* __restrict__ O, long n) {
  long i = ((long)blockIdx.x * blockDim.x + threadIdx.x) * 4;
  if (i + 4 <= n) {
    const float4 v = *(const float4*)(Y + i);
    u16x4 o;
    o.x = f2bf(v.x); o.y = f2bf(v.y); o.z = f2bf(v.z); o.w = f2bf(v.w);
    *(u16x4*)(O + i) = o;
  } else {
    for (long j = i; j < n; ++j) O[j] = f2bf(Y[j]);
  }
}

// ---------------------------------------------------------------------------
extern "C" void kernel_launch(void* const* d_in, const int* in_sizes, int n_in,
                              void* d_out, int out_size, void* d_ws, size_t ws_size,
                              hipStream_t stream) {
  const unsigned short* x   = (const unsigned short*)d_in[0];
  const float*          gw  = (const float*)d_in[1];
  const unsigned short* w1  = (const unsigned short*)d_in[2];
  const unsigned short* w2  = (const unsigned short*)d_in[3];
  const unsigned short* w3  = (const unsigned short*)d_in[4];
  const unsigned short* sw1 = (const unsigned short*)d_in[5];
  const unsigned short* sw2 = (const unsigned short*)d_in[6];
  const unsigned short* sw3 = (const unsigned short*)d_in[7];

  const int D  = in_sizes[1] / E_EXP;            // gate_w: [E, D]
  const int T  = in_sizes[0] / D;                // x: [B*S, D]
  const int I  = in_sizes[2] / (E_EXP * D);      // w1: [E, I, D]
  const int SI = in_sizes[5] / D;                // sw1: [SI, D]

  // Workspace carve-out (256B aligned regions).
  char* p = (char*)d_ws;
  auto carve = [&](size_t bytes) {
    char* r = p; p += (bytes + 255) & ~(size_t)255; return r;
  };
  int*   counts  = (int*)  carve(E_EXP * sizeof(int));
  int*   offsets = (int*)  carve(E_EXP * sizeof(int));
  int*   cursor  = (int*)  carve(E_EXP * sizeof(int));
  int*   top2i   = (int*)  carve((size_t)T * TOPK * sizeof(int));
  float* top2w   = (float*)carve((size_t)T * TOPK * sizeof(float));
  int*   tokidx  = (int*)  carve((size_t)T * TOPK * sizeof(int));
  float* tokw    = (float*)carve((size_t)T * TOPK * sizeof(float));
  float* yf      = (float*)carve((size_t)T * (size_t)D * sizeof(float));
  size_t helems  = (size_t)T * TOPK * (size_t)I;          // routed H slots
  const size_t hsh = (size_t)T * (size_t)SI;              // shared H
  if (hsh > helems) helems = hsh;
  unsigned short* Hbuf = (unsigned short*)carve(helems * sizeof(unsigned short));

  // 1) routing
  moe_init_kernel<<<1, 32, 0, stream>>>(counts);
  moe_gate_kernel<<<T, 128, 0, stream>>>(x, gw, top2i, top2w, counts, T, D);
  moe_scan_kernel<<<1, 1, 0, stream>>>(counts, offsets, cursor);
  moe_scatter_kernel<<<(T + 255) / 256, 256, 0, stream>>>(top2i, top2w, cursor,
                                                          tokidx, tokw, T);
  // 2) shared expert: H[T, SI] then Y[T, D] (direct store)
  moe_up_kernel<<<dim3(SI / TN, T / TM, 1), 256, 0, stream>>>(
      x, sw1, sw3, Hbuf, nullptr, nullptr, nullptr, T, SI, D, 0);
  moe_down_kernel<<<dim3(D / TN, T / TM, 1), 256, 0, stream>>>(
      Hbuf, sw2, yf, nullptr, nullptr, nullptr, nullptr, T, D, SI, 0);
  // 3) routed experts: grouped GEMMs, early-exit past per-expert counts
  moe_up_kernel<<<dim3(I / TN, T / TM, E_EXP), 256, 0, stream>>>(
      x, w1, w3, Hbuf, tokidx, counts, offsets, T, I, D, 1);
  moe_down_kernel<<<dim3(D / TN, T / TM, E_EXP), 256, 0, stream>>>(
      Hbuf, w2, yf, tokidx, tokw, counts, offsets, T, D, I, 1);
  // 4) convert to bf16
  const long n = (long)T * (long)D;
  moe_finalize_kernel<<<(unsigned)((n / 4 + 255) / 256), 256, 0, stream>>>(
      yf, (unsigned short*)d_out, n);
}
